// xModalKD_44066364457341
// MI455X (gfx1250) — compile-verified, matrix operand order checked by reading, not run on hardware
//
#include <hip/hip_runtime.h>
#include <math.h>

// ---------------------------------------------------------------------------
// Problem constants (from the reference)
// ---------------------------------------------------------------------------
#define S_  4
#define NV_ 80000
#define NP_ 120000
#define M_  60000
#define H_  256
#define C_  20
#define B_  2
#define HY_ 64
#define WX_ 1024

typedef __attribute__((ext_vector_type(16))) _Float16 v16h;
typedef __attribute__((ext_vector_type(8)))  _Float16 v8h;
typedef __attribute__((ext_vector_type(8)))  float    v8f;

// ---------------------------------------------------------------------------
// Fragment loader for v_wmma_f32_16x16x32_f16 (16-bit A/B VGPR layout).
// Caller pre-adds the lane's K-group offset g8 = (lane>=16)?8:0 into p, so
// both loads are immediate-offset 16-byte loads off one per-lane base.
// ---------------------------------------------------------------------------
__device__ __forceinline__ v16h ld_frag(const _Float16* p) {
  v8h a = *(const v8h*)(p);
  v8h b = *(const v8h*)(p + 16);
  v16h r;
#pragma unroll
  for (int i = 0; i < 8; ++i) { r[i] = a[i]; r[8 + i] = b[i]; }
  return r;
}

__device__ __forceinline__ v8f wmma_f16(v16h a, v16h b, v8f c) {
  return __builtin_amdgcn_wmma_f32_16x16x32_f16(false, a, false, b,
                                                (short)0, c, false, false);
}

__device__ __forceinline__ int imin(int a, int b) { return a < b ? a : b; }

// ---------------------------------------------------------------------------
// f32 -> f16 elementwise convert
// ---------------------------------------------------------------------------
__global__ void k_f32tof16(const float* __restrict__ src, _Float16* __restrict__ dst,
                           long long n) {
  long long i = (long long)blockIdx.x * blockDim.x + threadIdx.x;
  if (i < n) dst[i] = (_Float16)src[i];
}

// ---------------------------------------------------------------------------
// Weight convert + transpose:  W (Sn, K, N) f32  ->  Wt (Sn, N, K) f16
// ---------------------------------------------------------------------------
__global__ void k_wt(const float* __restrict__ W, _Float16* __restrict__ Wt,
                     int Sn, int K, int N) {
  long long total = (long long)Sn * K * N;
  long long i = (long long)blockIdx.x * blockDim.x + threadIdx.x;
  if (i >= total) return;
  int n = (int)(i % N);
  long long r = i / N;
  int k = (int)(r % K);
  int s = (int)(r / K);
  Wt[((long long)s * N + n) * K + k] = (_Float16)W[i];
}

// ---------------------------------------------------------------------------
// range_feats gather: out[s,p,h] = g[s, b_idx[p], h, y_idx[p], x_idx[p]]
// ---------------------------------------------------------------------------
__global__ void k_gather(const float* __restrict__ g,
                         const int* __restrict__ bI, const int* __restrict__ yI,
                         const int* __restrict__ xI,
                         _Float16* __restrict__ outR) {
  int p = blockIdx.x;
  int s = blockIdx.y;
  int h = threadIdx.x;              // H_ == 256 == blockDim.x
  long long plane = (long long)HY_ * WX_;
  long long base = (((long long)s * B_ + bI[p]) * H_) * plane
                 + (long long)yI[p] * WX_ + xI[p];
  float v = g[base + (long long)h * plane];
  outR[((long long)s * NP_ + p) * H_ + h] = (_Float16)v;
}

// ---------------------------------------------------------------------------
// Generic batched GEMM (K0/K1 compile-time so the K loop fully unrolls and
// every fragment load is an immediate-offset global_load_b128):
//   out[s, m, n] = act( concat(A0,A1)[s, m, :] @ W[s] + bias[s] )
//   Block = 256 thr = 8 waves; per-wave tile 16 rows x 128 cols
//   (8 v8f accumulators).  Per k-step: A fragment + all 8 B fragments are
//   loaded into distinct registers BEFORE the 8 WMMAs so the scheduler can
//   use partial loadcnt waits and overlap loads with matrix math.
//   Requires N % 128 == 0.
//   act: 0 = relu (f16), 1 = none (f16), 2 = relu(gate * sigmoid(x)) (f16)
// ---------------------------------------------------------------------------
template <int K0, int K1>
__global__ void k_gemm_t(const _Float16* __restrict__ A0, const int* __restrict__ idx0,
                         long long a0batch,
                         const _Float16* __restrict__ A1, long long a1batch,
                         const _Float16* __restrict__ Wt, long long wtbatch,
                         const float* __restrict__ bias,
                         const _Float16* __restrict__ gate, long long gatebatch,
                         _Float16* __restrict__ outp,
                         int Mrows, int N, int act) {
  constexpr int K = K0 + K1;
  int lane = threadIdx.x & 31;
  int wv   = threadIdx.x >> 5;
  int lm   = lane & 15;
  int g8   = (lane >> 4) * 8;        // K-group offset AND M-row offset (both 0/8)
  int s    = blockIdx.z;
  int n0   = blockIdx.x * 128;
  int rowBase = blockIdx.y * 128 + wv * 16;
  int row  = rowBase + lm;
  int rc   = imin(row, Mrows - 1);

  int sr = idx0 ? idx0[rc] : rc;
  const _Float16* a0row = A0 + (long long)s * a0batch + (long long)sr * K0 + g8;
  const _Float16* bbase = Wt + (long long)s * wtbatch
                        + (long long)(n0 + lm) * K + g8;

  v8f acc[8];
#pragma unroll
  for (int nt = 0; nt < 8; ++nt)
    acc[nt] = (v8f){0.f, 0.f, 0.f, 0.f, 0.f, 0.f, 0.f, 0.f};

  // phase 1: A0 columns [0, K0)
#pragma unroll
  for (int k0 = 0; k0 < K0; k0 += 32) {
    v16h a = ld_frag(a0row + k0);
    v16h b[8];
#pragma unroll
    for (int nt = 0; nt < 8; ++nt)
      b[nt] = ld_frag(bbase + nt * 16 * K + k0);
#pragma unroll
    for (int nt = 0; nt < 8; ++nt)
      acc[nt] = wmma_f16(a, b[nt], acc[nt]);
  }
  // phase 2: A1 columns [K0, K0+K1)
  if (K1 > 0) {
    const _Float16* a1row = A1 + (long long)s * a1batch + (long long)rc * K1 + g8;
#pragma unroll
    for (int k0 = 0; k0 < K1; k0 += 32) {
      v16h a = ld_frag(a1row + k0);
      v16h b[8];
#pragma unroll
      for (int nt = 0; nt < 8; ++nt)
        b[nt] = ld_frag(bbase + nt * 16 * K + K0 + k0);
#pragma unroll
      for (int nt = 0; nt < 8; ++nt)
        acc[nt] = wmma_f16(a, b[nt], acc[nt]);
    }
  }

#pragma unroll
  for (int nt = 0; nt < 8; ++nt) {
    int   nIdx = n0 + nt * 16 + lm;
    float bv   = bias ? bias[(long long)s * N + nIdx] : 0.f;
#pragma unroll
    for (int v = 0; v < 8; ++v) {
      int m = rowBase + v + g8;
      if (m >= Mrows) continue;
      float x = acc[nt][v] + bv;
      long long o = ((long long)s * Mrows + m) * N + nIdx;
      if (act == 0) {
        outp[o] = (_Float16)fmaxf(x, 0.f);
      } else if (act == 1) {
        outp[o] = (_Float16)x;
      } else {
        float gsig = 1.f / (1.f + __expf(-x));
        float fc = (float)gate[(long long)s * gatebatch + (long long)m * N + nIdx];
        outp[o] = (_Float16)fmaxf(fc * gsig, 0.f);
      }
    }
  }
}

// ---------------------------------------------------------------------------
// Fused 2-layer MLP head:  out = relu(A @ W1 + b1) @ W2 + b2  (hidden = 128)
//   A element (row, k) at: A + s*aBatch + row*rowStride + (k>>8)*segStride + (k&255)
//   (segStride = 256 -> contiguous; segStride = Mrows*H gives the (m, s*H+h)
//    transpose view for img_cat / pts_cat without materializing it.)
//   K compile-time -> fully unrolled, immediate-offset fragment loads,
//   B fragments preloaded per k-step into distinct registers.
//   Hidden 16x128 tile stashed in per-wave LDS, re-read in A-fragment layout.
// ---------------------------------------------------------------------------
template <int K>
__global__ void k_head2_t(const _Float16* __restrict__ A, long long aBatch,
                          long long rowStride, long long segStride,
                          const _Float16* __restrict__ W1t, long long w1b, // (128, K)
                          const float* __restrict__ B1, long long b1b,     // (128)
                          const _Float16* __restrict__ W2t, long long w2b, // (Cn, 128)
                          const float* __restrict__ B2, long long b2b,     // (Cn)
                          float* __restrict__ outp, long long outBatch,
                          int Mrows, int Cn) {
  __shared__ __align__(16) _Float16 hidden[8][16][128];   // 32 KB

  int lane = threadIdx.x & 31;
  int wv   = threadIdx.x >> 5;
  int lm   = lane & 15;
  int g8   = (lane >> 4) * 8;
  int s    = blockIdx.z;
  int rowBase = blockIdx.x * 128 + wv * 16;
  int row  = rowBase + lm;
  int rc   = imin(row, Mrows - 1);

  const _Float16* aRow = A + (long long)s * aBatch
                       + (long long)rc * rowStride + g8;
  long long segAdj = segStride - 256;   // addr = k0 + (k0>>8)*(segStride-256)

  // ---- layer 1: hidden[16 x 128], relu, stash in LDS ----
  {
    const _Float16* b1base = W1t + (long long)s * w1b + (long long)lm * K + g8;
    v8f acc[8];
#pragma unroll
    for (int nt = 0; nt < 8; ++nt)
      acc[nt] = (v8f){0.f, 0.f, 0.f, 0.f, 0.f, 0.f, 0.f, 0.f};

#pragma unroll
    for (int k0 = 0; k0 < K; k0 += 32) {
      v16h a = ld_frag(aRow + (long long)(k0 >> 8) * segAdj + k0);
      v16h b[8];
#pragma unroll
      for (int nt = 0; nt < 8; ++nt)
        b[nt] = ld_frag(b1base + nt * 16 * K + k0);
#pragma unroll
      for (int nt = 0; nt < 8; ++nt)
        acc[nt] = wmma_f16(a, b[nt], acc[nt]);
    }
#pragma unroll
    for (int nt = 0; nt < 8; ++nt) {
      float bv = B1[(long long)s * b1b + nt * 16 + lm];
#pragma unroll
      for (int v = 0; v < 8; ++v)
        hidden[wv][v + g8][nt * 16 + lm] = (_Float16)fmaxf(acc[nt][v] + bv, 0.f);
    }
  }
  __syncthreads();   // belt-and-braces; per-wave DS ordering already suffices

  // ---- layer 2: [16 x 128] @ [128 x Cn] -> logits ----
#pragma unroll
  for (int nt = 0; nt < 2; ++nt) {
    int ncol = nt * 16 + lm;
    int ncl  = imin(ncol, Cn - 1);
    const _Float16* brow = W2t + (long long)s * w2b + (long long)ncl * 128 + g8;
    const _Float16* ap   = &hidden[wv][lm][0] + g8;
    v8f acc = {0.f, 0.f, 0.f, 0.f, 0.f, 0.f, 0.f, 0.f};
#pragma unroll
    for (int k0 = 0; k0 < 128; k0 += 32) {
      v16h a = ld_frag(ap + k0);
      v16h b = ld_frag(brow + k0);
      acc = wmma_f16(a, b, acc);
    }
    float bv = B2[(long long)s * b2b + ncl];
#pragma unroll
    for (int v = 0; v < 8; ++v) {
      int m = rowBase + v + g8;
      if (m < Mrows && ncol < Cn)
        outp[(long long)s * outBatch + (long long)m * Cn + ncol] = acc[v] + bv;
    }
  }
}

// ---------------------------------------------------------------------------
// Host orchestration
// ---------------------------------------------------------------------------
extern "C" void kernel_launch(void* const* d_in, const int* in_sizes, int n_in,
                              void* d_out, int out_size, void* d_ws, size_t ws_size,
                              hipStream_t stream) {
  (void)in_sizes; (void)n_in; (void)out_size; (void)ws_size;

  const float* pts_feat   = (const float*)d_in[0];
  const float* pts_feat_f = (const float*)d_in[1];
  const float* g_img      = (const float*)d_in[2];
  const float* P[28];
  for (int i = 0; i < 28; ++i) P[i] = (const float*)d_in[3 + i];
  const int* b_idx = (const int*)d_in[31];
  const int* y_idx = (const int*)d_in[32];
  const int* x_idx = (const int*)d_in[33];
  const int* p2img = (const int*)d_in[34];
  float* out = (float*)d_out;

  // Output layout (concatenated in reference return order)
  long long o_img  = 0;
  long long o_pts  = (long long)M_ * C_;
  long long o_full = o_pts + (long long)NP_ * C_;
  long long o_fuse = o_full + (long long)S_ * NV_ * C_;

  // ---- carve workspace (all f16, 256B aligned) ----
  char* wsp = (char*)d_ws;
  auto carve = [&](size_t elems) -> _Float16* {
    _Float16* p = (_Float16*)wsp;
    wsp += ((elems * sizeof(_Float16) + 255) & ~(size_t)255);
    return p;
  };
  _Float16* pf16    = carve((size_t)S_ * NV_ * H_);
  _Float16* pff16   = carve((size_t)S_ * NP_ * H_);
  _Float16* rangeF  = carve((size_t)S_ * NP_ * H_);
  _Float16* flC     = carve((size_t)S_ * M_  * H_);
  _Float16* fcatC   = carve((size_t)S_ * M_  * H_);
  _Float16* fuseC   = carve((size_t)S_ * M_  * H_);
  _Float16* flL     = carve((size_t)S_ * NP_ * H_);
  _Float16* fcatL   = carve((size_t)S_ * NP_ * H_);
  _Float16* fuseL   = carve((size_t)S_ * NP_ * H_);
  _Float16* t_cls3d1 = carve((size_t)S_ * 128 * 256);
  _Float16* t_cls3d2 = carve((size_t)S_ * 20  * 128);
  _Float16* t_clsC1  = carve((size_t)S_ * 128 * 256);
  _Float16* t_clsC2  = carve((size_t)S_ * 20  * 128);
  _Float16* t_lnL2C  = carve((size_t)S_ * 256 * 256);
  _Float16* t_f1L2C  = carve((size_t)S_ * 256 * 512);
  _Float16* t_f2L2C  = carve((size_t)S_ * 256 * 256);
  _Float16* t_lnC2L  = carve((size_t)S_ * 256 * 256);
  _Float16* t_f1C2L  = carve((size_t)S_ * 256 * 512);
  _Float16* t_f2C2L  = carve((size_t)S_ * 256 * 256);
  _Float16* t_fcC1   = carve((size_t)128 * 1024);
  _Float16* t_fcC2   = carve((size_t)20  * 128);
  _Float16* t_fcL1   = carve((size_t)128 * 1024);
  _Float16* t_fcL2   = carve((size_t)20  * 128);

  auto cvt = [&](const float* src, _Float16* dst, long long n) {
    k_f32tof16<<<dim3((unsigned)((n + 255) / 256)), dim3(256), 0, stream>>>(src, dst, n);
  };
  auto wt = [&](const float* W, _Float16* Wt, int Sn, int K, int N) {
    long long n = (long long)Sn * K * N;
    k_wt<<<dim3((unsigned)((n + 255) / 256)), dim3(256), 0, stream>>>(W, Wt, Sn, K, N);
  };

  // ---- input conversions & weight transposes ----
  cvt(pts_feat,   pf16,  (long long)S_ * NV_ * H_);
  cvt(pts_feat_f, pff16, (long long)S_ * NP_ * H_);
  wt(P[0],  t_cls3d1, S_, 256, 128);
  wt(P[2],  t_cls3d2, S_, 128, 20);
  wt(P[4],  t_clsC1,  S_, 256, 128);
  wt(P[6],  t_clsC2,  S_, 128, 20);
  wt(P[8],  t_lnL2C,  S_, 256, 256);
  wt(P[10], t_f1L2C,  S_, 512, 256);
  wt(P[12], t_f2L2C,  S_, 256, 256);
  wt(P[14], t_lnC2L,  S_, 256, 256);
  wt(P[16], t_f1C2L,  S_, 512, 256);
  wt(P[18], t_f2C2L,  S_, 256, 256);
  wt(P[20], t_fcC1,   1, 1024, 128);
  wt(P[22], t_fcC2,   1, 128, 20);
  wt(P[24], t_fcL1,   1, 1024, 128);
  wt(P[26], t_fcL2,   1, 128, 20);

  // ---- range_feats gather ----
  k_gather<<<dim3(NP_, S_), dim3(H_), 0, stream>>>(g_img, b_idx, y_idx, x_idx, rangeF);

  dim3 blk(256);
  const long long npB = (long long)NP_ * H_;
  const long long mB  = (long long)M_ * H_;

  // ---- pts_pred_full head: relu(pts_feat@cls3d1)@cls3d2 -> d_out[o_full] ----
  k_head2_t<256><<<dim3((NV_ + 127) / 128, 1, S_), blk, 0, stream>>>(
      pf16, (long long)NV_ * H_, 256, 256,
      t_cls3d1, 128 * 256, P[1], 128,
      t_cls3d2, 20 * 128,  P[3], 20,
      out + o_full, (long long)NV_ * C_, NV_, C_);

  // ---- L2C branch (rows gathered by p2img_idx) ----
  // flC = relu( pts_feat_f[p2img] @ w_lnL2C + b )
  k_gemm_t<256, 0><<<dim3(2, (M_ + 127) / 128, S_), blk, 0, stream>>>(
      pff16, p2img, npB, nullptr, 0,
      t_lnL2C, 256 * 256, P[9], nullptr, 0, flC, M_, 256, 0);
  // fcatC = concat(rangeF[p2img], flC) @ w_f1L2C + b
  k_gemm_t<256, 256><<<dim3(2, (M_ + 127) / 128, S_), blk, 0, stream>>>(
      rangeF, p2img, npB, flC, mB,
      t_f1L2C, 256 * 512, P[11], nullptr, 0, fcatC, M_, 256, 1);
  // fuseC = relu( fcatC * sigmoid(fcatC @ w_f2L2C + b) )
  k_gemm_t<256, 0><<<dim3(2, (M_ + 127) / 128, S_), blk, 0, stream>>>(
      fcatC, nullptr, mB, nullptr, 0,
      t_f2L2C, 256 * 256, P[13], fcatC, mB, fuseC, M_, 256, 2);
  // fuse_pred_C head -> d_out[o_fuse]
  k_head2_t<256><<<dim3((M_ + 127) / 128, 1, S_), blk, 0, stream>>>(
      fuseC, mB, 256, 256,
      t_clsC1, 128 * 256, P[5], 128,
      t_clsC2, 20 * 128,  P[7], 20,
      out + o_fuse, (long long)M_ * C_, M_, C_);

  // ---- C2L branch (full NP rows) ----
  k_gemm_t<256, 0><<<dim3(2, (NP_ + 127) / 128, S_), blk, 0, stream>>>(
      rangeF, nullptr, npB, nullptr, 0,
      t_lnC2L, 256 * 256, P[15], nullptr, 0, flL, NP_, 256, 0);
  k_gemm_t<256, 256><<<dim3(2, (NP_ + 127) / 128, S_), blk, 0, stream>>>(
      pff16, nullptr, npB, flL, npB,
      t_f1C2L, 256 * 512, P[17], nullptr, 0, fcatL, NP_, 256, 1);
  k_gemm_t<256, 0><<<dim3(2, (NP_ + 127) / 128, S_), blk, 0, stream>>>(
      fcatL, nullptr, npB, nullptr, 0,
      t_f2C2L, 256 * 256, P[19], fcatL, npB, fuseL, NP_, 256, 2);

  // ---- img_seg head: A is (m, s*H+h) view of fuseC via segmented stride ----
  k_head2_t<1024><<<dim3((M_ + 127) / 128, 1, 1), blk, 0, stream>>>(
      fuseC, 0, 256, (long long)M_ * H_,
      t_fcC1, 0, P[21], 0,
      t_fcC2, 0, P[23], 0,
      out + o_img, 0, M_, C_);

  // ---- pts_seg head: same trick over fuseL ----
  k_head2_t<1024><<<dim3((NP_ + 127) / 128, 1, 1), blk, 0, stream>>>(
      fuseL, 0, 256, (long long)NP_ * H_,
      t_fcL1, 0, P[25], 0,
      t_fcL2, 0, P[27], 0,
      out + o_pts, 0, NP_, C_);
}
